// WindowedLocalMamba_56556129354513
// MI455X (gfx1250) — compile-verified, matrix operand order checked by reading, not run on hardware
//
#include <hip/hip_runtime.h>

typedef __attribute__((ext_vector_type(8)))  float  v8f;
typedef __attribute__((ext_vector_type(16))) __bf16 v16bf;
typedef unsigned short ush;

#define DM   256   // D_MODEL
#define DI   512   // D_INNER
#define DSZ  16    // D_STATE
#define DRK  16    // DT_RANK
#define WSZ  32    // window size (tokens)
#define WPB  2     // windows per block
#define MR   (WSZ * WPB)   // 64 rows per block
#define NTHREADS 256

// ---------------- bf16 hi/lo helpers ----------------
__device__ __forceinline__ void split_bf16(float x, ush& hi, ush& lo) {
  __bf16 h = (__bf16)x;                       // RNE
  float  r = x - (float)h;
  __bf16 l = (__bf16)r;
  hi = __builtin_bit_cast(ush, h);
  lo = __builtin_bit_cast(ush, l);
}
__device__ __forceinline__ void stage_pair(ush* sh, ush* sl, int idx, float x) {
  ush h, l; split_bf16(x, h, l);
  sh[idx] = h; sl[idx] = l;
}
__device__ __forceinline__ float unstage(const ush* sh, const ush* sl, int idx) {
  return (float)__builtin_bit_cast(__bf16, sh[idx]) +
         (float)__builtin_bit_cast(__bf16, sl[idx]);
}

union Frag { uint4 q[2]; v16bf v; };

// NS strips of 64x16 output: D[0:64][nt*16 ...] = A(64xK, LDS) * B^T, bf16x3 split-float.
//  - A fragments loaded once per chunk and shared by all NS strips (8 indep. WMMA chains
//    for NS=2 -> fills the 4-slot WMMA->VALU co-exec hazard window).
//  - Per-mi A base pointers hoisted out of the chunk loop so all DS/global offsets are
//    small immediates (no per-load address VALU).
template <int K, int NS>
__device__ __forceinline__ void mm_strips(const ush* __restrict__ sAhi,
                                          const ush* __restrict__ sAlo,
                                          const ush* __restrict__ Whi,
                                          const ush* __restrict__ Wlo,
                                          int nt0, int ntStride, int lane,
                                          v8f acc[NS][4]) {
  constexpr int KC = K / 32;
  const int lr = lane & 15, g = lane >> 4;
  const ush* amh[4]; const ush* aml[4];
#pragma unroll
  for (int mi = 0; mi < 4; ++mi) {
    amh[mi] = sAhi + (mi * 16 + lr) * K + 8 * g;
    aml[mi] = sAlo + (mi * 16 + lr) * K + 8 * g;
  }
  const ush* bh[NS]; const ush* bl[NS];
#pragma unroll
  for (int s = 0; s < NS; ++s) {
    const int nt = nt0 + s * ntStride;
    bh[s] = Whi + ((size_t)nt * KC * 32 + lane) * 16;
    bl[s] = Wlo + ((size_t)nt * KC * 32 + lane) * 16;
  }
#pragma unroll 1
  for (int c = 0; c < KC; ++c) {
    Frag bhf[NS], blf[NS];
#pragma unroll
    for (int s = 0; s < NS; ++s) {
      bhf[s].q[0] = *(const uint4*)(bh[s] + c * 512);      // 32B/lane, coalesced
      bhf[s].q[1] = *(const uint4*)(bh[s] + c * 512 + 8);
      blf[s].q[0] = *(const uint4*)(bl[s] + c * 512);
      blf[s].q[1] = *(const uint4*)(bl[s] + c * 512 + 8);
    }
#pragma unroll
    for (int mi = 0; mi < 4; ++mi) {
      Frag ah, al;
      ah.q[0] = *(const uint4*)(amh[mi] + c * 32);         // K = k0+8g..+7
      ah.q[1] = *(const uint4*)(amh[mi] + c * 32 + 16);    // K = k0+16+8g..+7
      al.q[0] = *(const uint4*)(aml[mi] + c * 32);
      al.q[1] = *(const uint4*)(aml[mi] + c * 32 + 16);
#pragma unroll
      for (int s = 0; s < NS; ++s) {
        acc[s][mi] = __builtin_amdgcn_wmma_f32_16x16x32_bf16(false, ah.v, false, bhf[s].v, (short)0, acc[s][mi], false, false);
        acc[s][mi] = __builtin_amdgcn_wmma_f32_16x16x32_bf16(false, ah.v, false, blf[s].v, (short)0, acc[s][mi], false, false);
        acc[s][mi] = __builtin_amdgcn_wmma_f32_16x16x32_bf16(false, al.v, false, bhf[s].v, (short)0, acc[s][mi], false, false);
      }
    }
  }
}

__device__ __forceinline__ float wave_sum(float v) {
#pragma unroll
  for (int off = 16; off > 0; off >>= 1) v += __shfl_xor(v, off, 32);
  return v;
}
__device__ __forceinline__ float fast_sigmoid(float x) { return 1.f / (1.f + __expf(-x)); }

// ---------------- weight prep: f32 [N][K] -> bf16 hi/lo in fragment order ----------------
__global__ void prep_weights_kernel(const float* __restrict__ W,
                                    ush* __restrict__ hi, ush* __restrict__ lo,
                                    int N, int K) {
  const int idx = blockIdx.x * NTHREADS + threadIdx.x;
  if (idx >= N * K) return;
  const int KC   = K / 32;
  const int e    = idx & 15;
  const int L    = (idx >> 4) & 31;
  const int blk  = idx >> 9;
  const int tile = blk / KC, chunk = blk % KC;
  const int g    = L >> 4, lr = L & 15;
  const int v    = e >> 1, pos = e & 1;
  const int kk   = (v < 4) ? (8 * g + 2 * v + pos) : (16 + 8 * g + 2 * (v - 4) + pos);
  const float w  = W[(size_t)(tile * 16 + lr) * K + chunk * 32 + kk];
  ush h, l; split_bf16(w, h, l);
  hi[idx] = h; lo[idx] = l;
}

// ---------------- fused 2-windows-per-block Mamba kernel ----------------
__global__ __launch_bounds__(NTHREADS)
void mamba_window_kernel(const float* __restrict__ x,
                         const float* __restrict__ in_b,
                         const float* __restrict__ ln_g,  const float* __restrict__ ln_b,
                         const float* __restrict__ conv_w, const float* __restrict__ conv_b,
                         const float* __restrict__ dtproj_w, const float* __restrict__ dtproj_b,
                         const float* __restrict__ A_log,  const float* __restrict__ Dp,
                         const float* __restrict__ out_b,
                         const float* __restrict__ oln_g,  const float* __restrict__ oln_b,
                         const ush* __restrict__ inw_h,  const ush* __restrict__ inw_l,
                         const ush* __restrict__ inpj_h, const ush* __restrict__ inpj_l,
                         const ush* __restrict__ xpj_h,  const ush* __restrict__ xpj_l,
                         const ush* __restrict__ opj_h,  const ush* __restrict__ opj_l,
                         const ush* __restrict__ outw_h, const ush* __restrict__ outw_l,
                         float* __restrict__ out) {
  extern __shared__ char smem_raw[];
  float* bufR = (float*)smem_raw;              // 64x256 f32 : residual -> final pre-LN
  float* bufP = bufR + MR * DM;                // 64x48  f32 : [dt | B | C] per token
  ush* hlnH = (ush*)(bufP + MR * 48);          // 64x256 bf16: h_ln, later h2
  ush* hlnL = hlnH + MR * DM;
  ush* genH = hlnL + MR * DM;                  // 64x512 bf16: x / xm / u->y / gated y
  ush* genL = genH + MR * DI;

  const int tid  = threadIdx.x;
  const int lane = tid & 31;
  const int wave = tid >> 5;
  const int lr   = lane & 15;
  const int grp  = lane >> 4;

  // ---- 1. load 2 windows of x, stage bf16 hi/lo (A of GEMM1; row stride DM) ----
  {
    const float* xg = x + (size_t)blockIdx.x * MR * DM;
#pragma unroll
    for (int i = tid; i < MR * DM; i += NTHREADS)
      stage_pair(genH, genL, i, xg[i]);
    __builtin_prefetch(inw_h + tid * 64, 0, 1);
  }
  __syncthreads();

  // ---- 2. GEMM1: h = x @ in_w^T + in_b -> bufR (residual) ----
  {
    v8f acc[2][4] = {};
    mm_strips<DM, 2>(genH, genL, inw_h, inw_l, wave, 8, lane, acc);
#pragma unroll
    for (int s = 0; s < 2; ++s) {
      const int n = (wave + s * 8) * 16 + lr;
      const float bias = in_b[n];
#pragma unroll
      for (int mi = 0; mi < 4; ++mi)
#pragma unroll
        for (int r = 0; r < 8; ++r)
          bufR[(mi * 16 + r + 8 * grp) * DM + n] = acc[s][mi][r] + bias;
    }
  }
  __syncthreads();

  // ---- 3. LayerNorm -> hln staging (bf16 hi/lo); 8 rows per wave ----
  for (int rr = 0; rr < 8; ++rr) {
    const int row = wave * 8 + rr;
    const float* p = bufR + row * DM;
    float s = 0.f;
    for (int j = lane; j < DM; j += 32) s += p[j];
    const float mu = wave_sum(s) * (1.f / DM);
    float vs = 0.f;
    for (int j = lane; j < DM; j += 32) { float d = p[j] - mu; vs += d * d; }
    const float rstd = rsqrtf(wave_sum(vs) * (1.f / DM) + 1e-5f);
    for (int j = lane; j < DM; j += 32)
      stage_pair(hlnH, hlnL, row * DM + j, (p[j] - mu) * rstd * ln_g[j] + ln_b[j]);
  }
  __syncthreads();

  // ---- 4. GEMM2a: xm = h_ln @ inproj_w[0:512]^T -> gen staging (bf16 hi/lo) ----
  for (int half = 0; half < 2; ++half) {
    const int nt0 = wave + half * 16;
    v8f acc[2][4] = {};
    mm_strips<DM, 2>(hlnH, hlnL, inpj_h, inpj_l, nt0, 8, lane, acc);
#pragma unroll
    for (int s = 0; s < 2; ++s) {
      const int n = (nt0 + s * 8) * 16 + lr;
#pragma unroll
      for (int mi = 0; mi < 4; ++mi)
#pragma unroll
        for (int r = 0; r < 8; ++r)
          stage_pair(genH, genL, (mi * 16 + r + 8 * grp) * DI + n, acc[s][mi][r]);
    }
  }
  __syncthreads();

  // ---- 5. depthwise causal conv (4 taps) + SiLU, in place in gen staging ----
  // thread owns channels {2*tid, 2*tid+1} for both windows -> in-place is race-free
#pragma unroll
  for (int cc = 0; cc < 2; ++cc) {
    const int d = tid * 2 + cc;
    const float w0 = conv_w[d * 4 + 0], w1 = conv_w[d * 4 + 1];
    const float w2 = conv_w[d * 4 + 2], w3 = conv_w[d * 4 + 3];
    const float cb = conv_b[d];
    for (int w = 0; w < WPB; ++w) {
      float h0 = 0.f, h1 = 0.f, h2 = 0.f;
      for (int t = 0; t < WSZ; ++t) {
        const int idx = (w * WSZ + t) * DI + d;
        const float cur = unstage(genH, genL, idx);
        float v = w0 * h0 + w1 * h1 + w2 * h2 + w3 * cur + cb;
        v = v * fast_sigmoid(v);
        stage_pair(genH, genL, idx, v);
        h0 = h1; h1 = h2; h2 = cur;
      }
    }
  }
  __syncthreads();

  // ---- 6. GEMM3: xdbc = xm @ xproj_w^T -> bufP (64 x 48) ----
  if (wave < 3) {
    v8f acc[1][4] = {};
    mm_strips<DI, 1>(genH, genL, xpj_h, xpj_l, wave, 0, lane, acc);
    const int n = wave * 16 + lr;
#pragma unroll
    for (int mi = 0; mi < 4; ++mi)
#pragma unroll
      for (int r = 0; r < 8; ++r)
        bufP[(mi * 16 + r + 8 * grp) * 48 + n] = acc[0][mi][r];
  }
  __syncthreads();

  // ---- 7. selective scan (window-sequential); params shared across windows ----
  {
    float Arow[2][DSZ], dtw[2][DRK], dtb2[2], Dp2[2];
#pragma unroll
    for (int cc = 0; cc < 2; ++cc) {
      const int d = tid * 2 + cc;
      dtb2[cc] = dtproj_b[d];
      Dp2[cc]  = Dp[d];
#pragma unroll
      for (int n = 0; n < DSZ; ++n) Arow[cc][n] = -__expf(A_log[d * DSZ + n]);
#pragma unroll
      for (int r = 0; r < DRK; ++r) dtw[cc][r] = dtproj_w[d * DRK + r];
    }
    for (int w = 0; w < WPB; ++w) {
      float hst[2][DSZ];
#pragma unroll
      for (int cc = 0; cc < 2; ++cc)
#pragma unroll
        for (int n = 0; n < DSZ; ++n) hst[cc][n] = 0.f;
      for (int t = 0; t < WSZ; ++t) {
        const int row = w * WSZ + t;
        float dtv[DRK], Bv[DSZ], Cv[DSZ];
#pragma unroll
        for (int i = 0; i < 16; ++i) {        // LDS broadcast reads
          dtv[i] = bufP[row * 48 + i];
          Bv[i]  = bufP[row * 48 + 16 + i];
          Cv[i]  = bufP[row * 48 + 32 + i];
        }
#pragma unroll
        for (int cc = 0; cc < 2; ++cc) {
          const int d = tid * 2 + cc;
          const int idx = row * DI + d;
          const float u = unstage(genH, genL, idx);
          float pre = dtb2[cc];
#pragma unroll
          for (int r = 0; r < DRK; ++r) pre += dtv[r] * dtw[cc][r];     // fused dtproj
          const float delta = (pre > 20.f) ? pre : log1pf(__expf(pre)); // softplus
          const float du = delta * u;
          float y = 0.f;
#pragma unroll
          for (int n = 0; n < DSZ; ++n) {
            const float hn = __expf(delta * Arow[cc][n]) * hst[cc][n] + du * Bv[n];
            hst[cc][n] = hn;
            y += hn * Cv[n];
          }
          stage_pair(genH, genL, idx, y + u * Dp2[cc]);  // y overwrites u in staging
        }
      }
    }
  }
  __syncthreads();

  // ---- 8. GEMM2b: z = h_ln @ inproj_w[512:1024]^T; gate y *= silu(z) in staging ----
  {
    const ush* zh = inpj_h + (size_t)32 * (DM / 32) * 512;  // z-half fragment blocks
    const ush* zl = inpj_l + (size_t)32 * (DM / 32) * 512;
    for (int half = 0; half < 2; ++half) {
      const int nt0 = wave + half * 16;
      v8f acc[2][4] = {};
      mm_strips<DM, 2>(hlnH, hlnL, zh, zl, nt0, 8, lane, acc);
#pragma unroll
      for (int s = 0; s < 2; ++s) {
        const int n = (nt0 + s * 8) * 16 + lr;
#pragma unroll
        for (int mi = 0; mi < 4; ++mi)
#pragma unroll
          for (int r = 0; r < 8; ++r) {
            const int idx = (mi * 16 + r + 8 * grp) * DI + n;
            const float z = acc[s][mi][r];
            stage_pair(genH, genL, idx, unstage(genH, genL, idx) * z * fast_sigmoid(z));
          }
      }
    }
  }
  __syncthreads();

  // ---- 9. GEMM5: h2 = y @ outproj_w^T + residual -> hln staging ----
  {
    v8f acc[2][4] = {};
    mm_strips<DI, 2>(genH, genL, opj_h, opj_l, wave, 8, lane, acc);
#pragma unroll
    for (int s = 0; s < 2; ++s) {
      const int n = (wave + s * 8) * 16 + lr;
#pragma unroll
      for (int mi = 0; mi < 4; ++mi)
#pragma unroll
        for (int r = 0; r < 8; ++r) {
          const int m = mi * 16 + r + 8 * grp;
          stage_pair(hlnH, hlnL, m * DM + n, acc[s][mi][r] + bufR[m * DM + n]);
        }
    }
  }
  __syncthreads();

  // ---- 10. GEMM6: pre-LN = h2 @ out_w^T + out_b -> bufR (f32) ----
  {
    v8f acc[2][4] = {};
    mm_strips<DM, 2>(hlnH, hlnL, outw_h, outw_l, wave, 8, lane, acc);
#pragma unroll
    for (int s = 0; s < 2; ++s) {
      const int n = (wave + s * 8) * 16 + lr;
      const float bias = out_b[n];
#pragma unroll
      for (int mi = 0; mi < 4; ++mi)
#pragma unroll
        for (int r = 0; r < 8; ++r)
          bufR[(mi * 16 + r + 8 * grp) * DM + n] = acc[s][mi][r] + bias;
    }
  }
  __syncthreads();

  // ---- 11. final LayerNorm -> global ----
  float* og = out + (size_t)blockIdx.x * MR * DM;
  for (int rr = 0; rr < 8; ++rr) {
    const int row = wave * 8 + rr;
    const float* p = bufR + row * DM;
    float s = 0.f;
    for (int j = lane; j < DM; j += 32) s += p[j];
    const float mu = wave_sum(s) * (1.f / DM);
    float vs = 0.f;
    for (int j = lane; j < DM; j += 32) { float d = p[j] - mu; vs += d * d; }
    const float rstd = rsqrtf(wave_sum(vs) * (1.f / DM) + 1e-5f);
    for (int j = lane; j < DM; j += 32)
      og[row * DM + j] = (p[j] - mu) * rstd * oln_g[j] + oln_b[j];
  }
}

extern "C" void kernel_launch(void* const* d_in, const int* in_sizes, int n_in,
                              void* d_out, int out_size, void* d_ws, size_t ws_size,
                              hipStream_t stream) {
  (void)n_in; (void)ws_size; (void)out_size;
  const float* x        = (const float*)d_in[0];
  const float* in_w     = (const float*)d_in[1];
  const float* in_b     = (const float*)d_in[2];
  const float* ln_g     = (const float*)d_in[3];
  const float* ln_b     = (const float*)d_in[4];
  const float* inproj_w = (const float*)d_in[5];
  const float* conv_w   = (const float*)d_in[6];
  const float* conv_b   = (const float*)d_in[7];
  const float* xproj_w  = (const float*)d_in[8];
  const float* dtproj_w = (const float*)d_in[9];
  const float* dtproj_b = (const float*)d_in[10];
  const float* A_log    = (const float*)d_in[11];
  const float* Dp       = (const float*)d_in[12];
  const float* outproj_w= (const float*)d_in[13];
  const float* out_w    = (const float*)d_in[14];
  const float* out_b    = (const float*)d_in[15];
  const float* oln_g    = (const float*)d_in[16];
  const float* oln_b    = (const float*)d_in[17];
  float* out = (float*)d_out;

  // ---- workspace: bf16 hi/lo fragment-order weights ----
  ush* ws = (ush*)d_ws;
  ush* inw_h  = ws;                 ush* inw_l  = inw_h  + 256 * 256;
  ush* inpj_h = inw_l  + 256 * 256; ush* inpj_l = inpj_h + 1024 * 256;
  ush* xpj_h  = inpj_l + 1024 * 256; ush* xpj_l = xpj_h  + 48 * 512;
  ush* opj_h  = xpj_l  + 48 * 512;  ush* opj_l  = opj_h  + 256 * 512;
  ush* outw_h = opj_l  + 256 * 512; ush* outw_l = outw_h + 256 * 256;

  prep_weights_kernel<<<(256 * 256) / NTHREADS, NTHREADS, 0, stream>>>(in_w, inw_h, inw_l, 256, 256);
  prep_weights_kernel<<<(1024 * 256) / NTHREADS, NTHREADS, 0, stream>>>(inproj_w, inpj_h, inpj_l, 1024, 256);
  prep_weights_kernel<<<(48 * 512) / NTHREADS, NTHREADS, 0, stream>>>(xproj_w, xpj_h, xpj_l, 48, 512);
  prep_weights_kernel<<<(256 * 512) / NTHREADS, NTHREADS, 0, stream>>>(outproj_w, opj_h, opj_l, 256, 512);
  prep_weights_kernel<<<(256 * 256) / NTHREADS, NTHREADS, 0, stream>>>(out_w, outw_h, outw_l, 256, 256);

  const int tokens = in_sizes[0] / DM;        // B*T = 32768
  const int blocks = tokens / (WSZ * WPB);    // 512
  const size_t smem = (size_t)(MR * DM + MR * 48) * sizeof(float)
                    + (size_t)(2 * MR * DM + 2 * MR * DI) * sizeof(ush); // 268 KB

  mamba_window_kernel<<<blocks, NTHREADS, smem, stream>>>(
      x, in_b, ln_g, ln_b, conv_w, conv_b, dtproj_w, dtproj_b, A_log, Dp,
      out_b, oln_g, oln_b,
      inw_h, inw_l, inpj_h, inpj_l, xpj_h, xpj_l, opj_h, opj_l, outw_h, outw_l,
      out);
}
